// MSAColAttention_81655918231691
// MI455X (gfx1250) — matrix operand order repaired and checked by reading, not compiled, not run on hardware
//
#include <hip/hip_runtime.h>

// MSA column attention, fused per residue-row r.  All GEMMs via
// v_wmma_f32_16x16x32_f16 (f16 inputs, f32 accumulate).
// N=1, S=256, R=384, M=256, HEAD=8, C=32.

typedef __attribute__((ext_vector_type(16))) _Float16 v16h;
typedef __attribute__((ext_vector_type(8)))  _Float16 v8h;
typedef __attribute__((ext_vector_type(8)))  float    v8f;

#define S_DIM 256
#define R_DIM 384
#define M_DIM 256
#define HEADS 8
#define CH    32
#define QK_SCALE 0.17677669529663687f

#define NWAVE 16
#define NTHREADS (NWAVE * 32)

// LDS layout (bytes)
#define XH_OFF   0                           // x_r as f16 [256][256]
#define ATT_OFF  (XH_OFF  + 256 * 256 * 2)   // gated attn out f16 [256][256]
#define KH_OFF   (ATT_OFF + 256 * 256 * 2)   // K_h f16 [256][32]   ([n=j][k=c])
#define VHT_OFF  (KH_OFF  + 256 * 32 * 2)    // V_h^T f16 [32][256] ([n=c][k=j])
#define SCR_OFF  (VHT_OFF + 32 * 256 * 2)    // per-wave 16x32 f16 tile scratch
#define LDS_BYTES (SCR_OFF + NWAVE * 16 * 32 * 2)   // 311296 B <= 320 KB

// ---------------- WMMA fragment helpers ----------------

__device__ __forceinline__ v8f wmma16(v16h a, v16h b, v8f c) {
  return __builtin_amdgcn_wmma_f32_16x16x32_f16(false, a, false, b, (short)0, c,
                                                false, false);
}

// A-matrix 16x32 f16 fragment from row-major [rows][ld] storage.
__device__ __forceinline__ v16h load_fragA(const _Float16* p, int row0, int k0,
                                           int ld) {
  const int lane = threadIdx.x & 31;
  const int m = lane & 15, hi = lane >> 4;
  const _Float16* base = p + (row0 + m) * ld + k0 + hi * 8;
  v8h c0 = *(const v8h*)(base);
  v8h c1 = *(const v8h*)(base + 16);
  v16h f;
#pragma unroll
  for (int i = 0; i < 8; ++i) { f[i] = c0[i]; f[i + 8] = c1[i]; }
  return f;
}

// B-matrix 32x16 f16 fragment from B^T stored row-major as [n][k] with stride ld.
__device__ __forceinline__ v16h load_fragB(const _Float16* p, int n0, int k0,
                                           int ld) {
  const int lane = threadIdx.x & 31;
  const int n = lane & 15, hi = lane >> 4;
  const _Float16* base = p + (n0 + n) * ld + k0 + hi * 8;
  v8h c0 = *(const v8h*)(base);
  v8h c1 = *(const v8h*)(base + 16);
  v16h f;
#pragma unroll
  for (int i = 0; i < 8; ++i) { f[i] = c0[i]; f[i + 8] = c1[i]; }
  return f;
}

// D fragment (16x16 f32): element (M = r + 8*(lane>=16), N = lane&15) in d[r].
__device__ __forceinline__ void store_d_f16(v8f d, _Float16* p, int row0,
                                            int col0, int ld) {
  const int lane = threadIdx.x & 31;
  const int n = lane & 15, hi = lane >> 4;
#pragma unroll
  for (int r = 0; r < 8; ++r)
    p[(row0 + r + 8 * hi) * ld + col0 + n] = (_Float16)d[r];
}

// Transposed D store: (M=j, N=c) -> p[(col0+N)*ldT + row0+M]  (for V^T)
__device__ __forceinline__ void store_d_f16_T(v8f d, _Float16* p, int row0,
                                              int col0, int ldT) {
  const int lane = threadIdx.x & 31;
  const int n = lane & 15, hi = lane >> 4;
#pragma unroll
  for (int r = 0; r < 8; ++r)
    p[(col0 + n) * ldT + row0 + r + 8 * hi] = (_Float16)d[r];
}

// Row-max across a 16-lane half of the wave (D-matrix row lives in 16 lanes).
__device__ __forceinline__ float rmax16(float v) {
#pragma unroll
  for (int m = 1; m < 16; m <<= 1) v = fmaxf(v, __shfl_xor(v, m, 32));
  return v;
}

// ---------------- kernels ----------------

// Transpose + f16-convert weights into B^T ([n][k]) layout.
__global__ void __launch_bounds__(256)
prep_weights(const float* __restrict__ gate_proj,
             const float* __restrict__ qkv_proj,
             const float* __restrict__ out_proj,
             _Float16* __restrict__ wqT, _Float16* __restrict__ wkT,
             _Float16* __restrict__ wvT, _Float16* __restrict__ wgT,
             _Float16* __restrict__ woT) {
  const int n = threadIdx.x;       // output column
  const int k = blockIdx.x;        // input row
  wgT[n * 256 + k] = (_Float16)gate_proj[k * 256 + n];
  wqT[n * 256 + k] = (_Float16)qkv_proj[k * 768 + n];
  wkT[n * 256 + k] = (_Float16)qkv_proj[k * 768 + 256 + n];
  wvT[n * 256 + k] = (_Float16)qkv_proj[k * 768 + 512 + n];
  woT[n * 256 + k] = (_Float16)out_proj[k * 256 + n];
}

// One workgroup per residue row r.  16 waves / WGP -> hard 256-VGPR budget
// per wave; k-loops use unroll 2 so the live fragment set stays within it.
__global__ void __launch_bounds__(NTHREADS, 1)
msa_col_attn(const float* __restrict__ msa,
             const _Float16* __restrict__ wqT, const _Float16* __restrict__ wkT,
             const _Float16* __restrict__ wvT, const _Float16* __restrict__ wgT,
             const _Float16* __restrict__ woT, float* __restrict__ out) {
  extern __shared__ char smem[];
  _Float16* xh  = (_Float16*)(smem + XH_OFF);
  _Float16* att = (_Float16*)(smem + ATT_OFF);
  _Float16* kh  = (_Float16*)(smem + KH_OFF);
  _Float16* vhT = (_Float16*)(smem + VHT_OFF);

  const int tid  = threadIdx.x;
  const int wave = tid >> 5;
  const int lane = tid & 31;
  const int rb   = blockIdx.x;     // residue row r
  _Float16* scr = (_Float16*)(smem + SCR_OFF) + wave * (16 * 32);

  // Constant all-ones B fragment: P x ones gives softmax row sums via WMMA.
  v16h bOnes;
#pragma unroll
  for (int i = 0; i < 16; ++i) bOnes[i] = (_Float16)1.0f;

  // ---- Phase 0: stage x_r = msa[:, rb, :] into LDS as f16 [256 s][256 m]
  {
    const int s = tid >> 1, half = tid & 1;
    const float4* src =
        (const float4*)(msa + ((size_t)s * R_DIM + rb) * M_DIM + half * 128);
    _Float16* dst = xh + s * 256 + half * 128;
#pragma unroll 4
    for (int i = 0; i < 32; ++i) {
      float4 v = src[i];
      dst[i * 4 + 0] = (_Float16)v.x;
      dst[i * 4 + 1] = (_Float16)v.y;
      dst[i * 4 + 2] = (_Float16)v.z;
      dst[i * 4 + 3] = (_Float16)v.w;
    }
  }
  __syncthreads();

  const int i0 = wave * 16;        // this wave's query-row tile

  for (int h = 0; h < HEADS; ++h) {
    const _Float16* wq = wqT + (size_t)h * CH * 256;
    const _Float16* wk = wkT + (size_t)h * CH * 256;
    const _Float16* wv = wvT + (size_t)h * CH * 256;
    const _Float16* wg = wgT + (size_t)h * CH * 256;

    // ---- K_h / V_h projections (cooperative: wave owns j-tile = wave).
    // K and V share the A fragment -> one LDS read feeds two WMMAs.
    {
      const int jt = wave;
#pragma unroll
      for (int nt = 0; nt < 2; ++nt) {
        v8f accK = {0.f, 0.f, 0.f, 0.f, 0.f, 0.f, 0.f, 0.f};
        v8f accV = accK;
#pragma unroll 2
        for (int k0 = 0; k0 < M_DIM; k0 += 32) {
          v16h a = load_fragA(xh, jt * 16, k0, 256);
          accK = wmma16(a, load_fragB(wk, nt * 16, k0, 256), accK);
          accV = wmma16(a, load_fragB(wv, nt * 16, k0, 256), accV);
        }
        store_d_f16(accK, kh, jt * 16, nt * 16, 32);
        store_d_f16_T(accV, vhT, jt * 16, nt * 16, 256);  // V^T [c][j]
      }
    }

    // ---- Q (scaled) and gate logits for this wave's i-tile (shared A frag)
    v8f gd0 = {0.f, 0.f, 0.f, 0.f, 0.f, 0.f, 0.f, 0.f};
    v8f gd1 = gd0;
    {
      v8f q0 = gd0, q1 = gd0;
#pragma unroll 2
      for (int k0 = 0; k0 < M_DIM; k0 += 32) {
        v16h a = load_fragA(xh, i0, k0, 256);
        q0 = wmma16(a, load_fragB(wq, 0, k0, 256), q0);
        q1 = wmma16(a, load_fragB(wq, 16, k0, 256), q1);
        gd0 = wmma16(a, load_fragB(wg, 0, k0, 256), gd0);
        gd1 = wmma16(a, load_fragB(wg, 16, k0, 256), gd1);
      }
#pragma unroll
      for (int r = 0; r < 8; ++r) { q0[r] *= QK_SCALE; q1[r] *= QK_SCALE; }
      store_d_f16(q0, scr, 0, 0, 32);
      store_d_f16(q1, scr, 0, 16, 32);
    }
    __syncthreads();  // kh/vhT fully written by all waves

    v16h qA = load_fragA(scr, 0, 0, 32);  // Q as A-fragment [16 x 32]

    // ---- flash attention over j (online softmax), 32 keys per step
    v8f o0 = {0.f, 0.f, 0.f, 0.f, 0.f, 0.f, 0.f, 0.f};
    v8f o1 = o0;
    float mrow[8], lrow[8];
#pragma unroll
    for (int r = 0; r < 8; ++r) { mrow[r] = -3.0e38f; lrow[r] = 0.f; }

#pragma unroll 1
    for (int j0 = 0; j0 < S_DIM; j0 += 32) {
      v16h bk0 = load_fragB(kh, j0, 0, 32);
      v16h bk1 = load_fragB(kh, j0 + 16, 0, 32);
      v8f z = {0.f, 0.f, 0.f, 0.f, 0.f, 0.f, 0.f, 0.f};
      v8f s1 = wmma16(qA, bk0, z);
      v8f s2 = wmma16(qA, bk1, z);
#pragma unroll
      for (int r = 0; r < 8; ++r) {
        float smax  = rmax16(fmaxf(s1[r], s2[r]));
        float mnew  = fmaxf(mrow[r], smax);
        float alpha = __expf(mrow[r] - mnew);
        float p1    = __expf(s1[r] - mnew);
        float p2    = __expf(s2[r] - mnew);
        lrow[r] *= alpha;        // row-sum contribution added via WMMA below
        mrow[r] = mnew;
        o0[r] *= alpha;
        o1[r] *= alpha;
        const int m = r + 8 * (lane >> 4), n = lane & 15;
        scr[m * 32 + n]      = (_Float16)p1;
        scr[m * 32 + 16 + n] = (_Float16)p2;
      }
      v16h pA  = load_fragA(scr, 0, 0, 32);
      v8f sums = wmma16(pA, bOnes, z);   // row sums of P, replicated per row
      v16h bv0 = load_fragB(vhT, 0, j0, 256);
      v16h bv1 = load_fragB(vhT, 16, j0, 256);
      o0 = wmma16(pA, bv0, o0);
      o1 = wmma16(pA, bv1, o1);
#pragma unroll
      for (int r = 0; r < 8; ++r) lrow[r] += sums[r];
    }

    // ---- normalize, gate (sigmoid), write to att LDS
#pragma unroll
    for (int r = 0; r < 8; ++r) {
      float inv = 1.f / lrow[r];
      float g0  = 1.f / (1.f + __expf(-gd0[r]));
      float g1  = 1.f / (1.f + __expf(-gd1[r]));
      const int m = i0 + r + 8 * (lane >> 4), n = lane & 15;
      att[m * 256 + h * CH + n]      = (_Float16)(o0[r] * inv * g0);
      att[m * 256 + h * CH + 16 + n] = (_Float16)(o1[r] * inv * g1);
    }
    __syncthreads();  // done with kh/vhT before next head overwrites
  }

  // ---- Phase 4: out_r = att @ out_proj, write f32 transposed to [S,R,M]
#pragma unroll 1
  for (int n0 = 0; n0 < M_DIM; n0 += 16) {
    v8f acc = {0.f, 0.f, 0.f, 0.f, 0.f, 0.f, 0.f, 0.f};
#pragma unroll 2
    for (int k0 = 0; k0 < M_DIM; k0 += 32) {
      v16h a = load_fragA(att, i0, k0, 256);
      v16h b = load_fragB(woT, n0, k0, 256);
      acc = wmma16(a, b, acc);
    }
    const int n = lane & 15, hi = lane >> 4;
#pragma unroll
    for (int r = 0; r < 8; ++r) {
      const int s = i0 + r + 8 * hi;
      out[((size_t)s * R_DIM + rb) * M_DIM + n0 + n] = acc[r];
    }
  }
}

// ---------------- launch ----------------

extern "C" void kernel_launch(void* const* d_in, const int* in_sizes, int n_in,
                              void* d_out, int out_size, void* d_ws,
                              size_t ws_size, hipStream_t stream) {
  (void)in_sizes; (void)n_in; (void)out_size; (void)ws_size;
  const float* msa       = (const float*)d_in[0];
  const float* gate_proj = (const float*)d_in[1];
  const float* qkv_proj  = (const float*)d_in[2];
  const float* out_proj  = (const float*)d_in[3];
  float* out = (float*)d_out;

  _Float16* wqT = (_Float16*)d_ws;
  _Float16* wkT = wqT + 256 * 256;
  _Float16* wvT = wkT + 256 * 256;
  _Float16* wgT = wvT + 256 * 256;
  _Float16* woT = wgT + 256 * 256;

  prep_weights<<<256, 256, 0, stream>>>(gate_proj, qkv_proj, out_proj, wqT, wkT,
                                        wvT, wgT, woT);

  (void)hipFuncSetAttribute((const void*)msa_col_attn,
                            hipFuncAttributeMaxDynamicSharedMemorySize,
                            (int)LDS_BYTES);
  msa_col_attn<<<R_DIM, NTHREADS, LDS_BYTES, stream>>>(msa, wqT, wkT, wvT, wgT,
                                                       woT, out);
}
